// GlobalAGCN_44109314130607
// MI455X (gfx1250) — matrix operand order repaired and checked by reading, not compiled, not run on hardware
//
#include <hip/hip_runtime.h>
#include <hip/hip_bf16.h>

typedef __bf16 bf16_t;
typedef __attribute__((ext_vector_type(16))) __bf16 bf16x16;
typedef __attribute__((ext_vector_type(8)))  __bf16 bf16x8;
typedef __attribute__((ext_vector_type(8)))  float  f32x8;

#define NNODES 4096
#define NBATCH 64
#define DIM    64
#define EMBED  10

// ---------------------------------------------------------------------------
// Kernel A: x [B, M, C] fp32  ->  XT [j = b*64+c][m] bf16  (tiled transpose)
// ---------------------------------------------------------------------------
__global__ __launch_bounds__(256) void k_build_xt(const float* __restrict__ x,
                                                  bf16_t* __restrict__ xt) {
  __shared__ __align__(16) bf16_t tile[64][72];  // padded to dodge bank conflicts
  const int b  = blockIdx.x;        // 0..63
  const int m0 = blockIdx.y * 64;   // m tile
  const int tid = threadIdx.x;
  #pragma unroll
  for (int i = 0; i < 16; ++i) {
    int idx = tid + i * 256;        // 0..4095
    int mm = idx >> 6, c = idx & 63;
    tile[mm][c] = (bf16_t)x[(size_t)b * NNODES * DIM + (size_t)(m0 + mm) * DIM + c];
  }
  __syncthreads();
  #pragma unroll
  for (int i = 0; i < 16; ++i) {
    int idx = tid + i * 256;
    int c = idx >> 6, mm = idx & 63;
    xt[(size_t)(b * 64 + c) * NNODES + m0 + mm] = tile[mm][c];
  }
}

// ---------------------------------------------------------------------------
// Kernel B: S[n,:] = softmax(relu(E[n] . E[m]))  -> bf16
// ---------------------------------------------------------------------------
__global__ __launch_bounds__(256) void k_softmax(const float* __restrict__ E,
                                                 bf16_t* __restrict__ S) {
  __shared__ float row[NNODES];
  __shared__ float red[256];
  const int n = blockIdx.x, tid = threadIdx.x;
  float e[EMBED];
  #pragma unroll
  for (int d = 0; d < EMBED; ++d) e[d] = E[n * EMBED + d];

  float lmax = 0.0f;  // relu output >= 0 and diag >= 0, so 0 is a valid floor
  for (int m = tid; m < NNODES; m += 256) {
    float dot = 0.f;
    #pragma unroll
    for (int d = 0; d < EMBED; ++d) dot += e[d] * E[m * EMBED + d];
    float v = fmaxf(dot, 0.f);
    row[m] = v;
    lmax = fmaxf(lmax, v);
  }
  red[tid] = lmax; __syncthreads();
  for (int s = 128; s > 0; s >>= 1) {
    if (tid < s) red[tid] = fmaxf(red[tid], red[tid + s]);
    __syncthreads();
  }
  const float smax = red[0]; __syncthreads();

  float lsum = 0.f;
  for (int m = tid; m < NNODES; m += 256) {
    float ex = __expf(row[m] - smax);
    row[m] = ex; lsum += ex;
  }
  red[tid] = lsum; __syncthreads();
  for (int s = 128; s > 0; s >>= 1) {
    if (tid < s) red[tid] += red[tid + s];
    __syncthreads();
  }
  const float inv = 1.0f / red[0];
  for (int m = tid; m < NNODES; m += 256)
    S[(size_t)n * NNODES + m] = (bf16_t)(row[m] * inv);
}

// ---------------------------------------------------------------------------
// Kernel C: C1[4096,4096] = S @ XT^T  (bf16 x bf16 -> fp32 acc, store bf16)
// Block tile 128x128, 8 waves, each wave 32x64 (2x4 WMMA tiles), K-step 32.
// Tiles staged with GLOBAL_LOAD_ASYNC_TO_LDS_B128 (ASYNCcnt), double-buffered.
// ---------------------------------------------------------------------------
__global__ __launch_bounds__(256) void k_diffuse(const bf16_t* __restrict__ S,
                                                 const bf16_t* __restrict__ XT,
                                                 bf16_t* __restrict__ C1) {
  __shared__ __align__(16) bf16_t As[2][128 * 32];  // [buf][row][k]
  __shared__ __align__(16) bf16_t Bs[2][128 * 32];  // [buf][col][k]
  const int tid  = threadIdx.x;
  const int lane = tid & 31;
  const int wave = tid >> 5;
  const int Mbase = blockIdx.y * 128;
  const int Nbase = blockIdx.x * 128;
  const int wm = wave & 3;   // 4 x 32-row slices
  const int wn = wave >> 2;  // 2 x 64-col slices

  const int hs   = (lane < 16) ? 0 : 1;
  const int l16  = lane & 15;
  const int a_k0 = hs * 16;   // A lanes 0-15: K 0-15; lanes 16-31: K 16-31
  const int b_k0 = hs * 16;   // B lanes 0-15: K 0-15; lanes 16-31: K 16-31

  // Each thread owns two 16B chunks per tile (512 chunks of 8 bf16 per 128x32 tile)
  const int q0 = tid, q1 = tid + 256;
  const int r0 = q0 >> 2, off0 = (q0 & 3) * 8;
  const int r1 = q1 >> 2, off1 = (q1 & 3) * 8;
  const bf16_t* gA0 = S  + (size_t)(Mbase + r0) * NNODES + off0;
  const bf16_t* gA1 = S  + (size_t)(Mbase + r1) * NNODES + off1;
  const bf16_t* gB0 = XT + (size_t)(Nbase + r0) * NNODES + off0;
  const bf16_t* gB1 = XT + (size_t)(Nbase + r1) * NNODES + off1;
  // LDS byte offsets (flat shared pointer low 32 bits == wave-relative LDS addr)
  unsigned lA0[2], lA1[2], lB0[2], lB1[2];
  #pragma unroll
  for (int bb = 0; bb < 2; ++bb) {
    lA0[bb] = (unsigned)(uintptr_t)&As[bb][r0 * 32 + off0];
    lA1[bb] = (unsigned)(uintptr_t)&As[bb][r1 * 32 + off1];
    lB0[bb] = (unsigned)(uintptr_t)&Bs[bb][r0 * 32 + off0];
    lB1[bb] = (unsigned)(uintptr_t)&Bs[bb][r1 * 32 + off1];
  }

  // 4 async DMA instructions per wave per K-step = 2 full 8KB tiles per block
  auto issue = [&](int buf, int kk) {
    asm volatile("global_load_async_to_lds_b128 %0, %1, off"
                 :: "v"(lA0[buf]), "v"((unsigned long long)(gA0 + kk)) : "memory");
    asm volatile("global_load_async_to_lds_b128 %0, %1, off"
                 :: "v"(lA1[buf]), "v"((unsigned long long)(gA1 + kk)) : "memory");
    asm volatile("global_load_async_to_lds_b128 %0, %1, off"
                 :: "v"(lB0[buf]), "v"((unsigned long long)(gB0 + kk)) : "memory");
    asm volatile("global_load_async_to_lds_b128 %0, %1, off"
                 :: "v"(lB1[buf]), "v"((unsigned long long)(gB1 + kk)) : "memory");
  };

  f32x8 acc[2][4];
  {
    f32x8 z = {};
    #pragma unroll
    for (int i = 0; i < 2; ++i)
      #pragma unroll
      for (int j = 0; j < 4; ++j) acc[i][j] = z;
  }

  issue(0, 0);  // prologue: fill buffer 0

  for (int kk = 0; kk < NNODES; kk += 32) {
    const int cur = (kk >> 5) & 1;
    const bool has_next = (kk + 32) < NNODES;
    if (has_next) {
      issue(cur ^ 1, kk + 32);  // buffer cur^1 safe: trailing barrier of prev iter
      // 8 outstanding; oldest 4 (= buffer cur) complete in order
      asm volatile("s_wait_asynccnt 0x4" ::: "memory");
    } else {
      asm volatile("s_wait_asynccnt 0x0" ::: "memory");
    }
    __syncthreads();

    bf16x16 a[2], b[4];
    #pragma unroll
    for (int mi = 0; mi < 2; ++mi) {
      int row = wm * 32 + mi * 16 + l16;
      const bf16_t* p = &As[cur][row * 32 + a_k0];
      bf16x8 lo = *(const bf16x8*)p;
      bf16x8 hi = *(const bf16x8*)(p + 8);
      #pragma unroll
      for (int t = 0; t < 8; ++t) { a[mi][t] = lo[t]; a[mi][t + 8] = hi[t]; }
    }
    #pragma unroll
    for (int ni = 0; ni < 4; ++ni) {
      int col = wn * 64 + ni * 16 + l16;
      b[ni] = *(const bf16x16*)(&Bs[cur][col * 32 + b_k0]);
    }
    #pragma unroll
    for (int mi = 0; mi < 2; ++mi)
      #pragma unroll
      for (int ni = 0; ni < 4; ++ni)
        acc[mi][ni] = __builtin_amdgcn_wmma_f32_16x16x32_bf16(
            false, a[mi], false, b[ni], (short)0, acc[mi][ni], false, false);
    __syncthreads();  // protects buffer cur before it is re-filled next+1 iter
  }

  // C/D layout: VGPR r -> M = r + 8*hs (per lane half); N = lane%16
  #pragma unroll
  for (int mi = 0; mi < 2; ++mi)
    #pragma unroll
    for (int ni = 0; ni < 4; ++ni) {
      int col = Nbase + wn * 64 + ni * 16 + l16;
      #pragma unroll
      for (int r = 0; r < 8; ++r) {
        int row = Mbase + wm * 32 + mi * 16 + r + hs * 8;
        C1[(size_t)row * NNODES + col] = (bf16_t)acc[mi][ni][r];
      }
    }
}

// ---------------------------------------------------------------------------
// Kernel D: per node n — build Wn[128][64] & bias from E, A = [x | C1-row],
// then [64x128] @ [128x64] WMMA GEMM + bias -> out[b,n,o] fp32.
// ---------------------------------------------------------------------------
__global__ __launch_bounds__(256) void k_final(const float* __restrict__ x,
                                               const float* __restrict__ E,
                                               const float* __restrict__ Wp,
                                               const float* __restrict__ Bp,
                                               const bf16_t* __restrict__ C1,
                                               float* __restrict__ out) {
  __shared__ __align__(16) bf16_t At[64 * 128];  // [b][ki]
  __shared__ __align__(16) bf16_t Wt[64 * 128];  // [o][ki] (B layout: K contiguous)
  __shared__ float biasv[64];
  const int n = blockIdx.x;
  const int tid = threadIdx.x;

  float e[EMBED];
  #pragma unroll
  for (int d = 0; d < EMBED; ++d) e[d] = E[n * EMBED + d];

  #pragma unroll
  for (int i = 0; i < 32; ++i) {            // Wn[ki][o] = sum_d e[d]*Wp[d][ki][o]
    int q  = tid + i * 256;                 // 0..8191
    int o  = q >> 7;
    int ki = q & 127;
    float s = 0.f;
    #pragma unroll
    for (int d = 0; d < EMBED; ++d) s += e[d] * Wp[d * (2 * 64 * 64) + ki * 64 + o];
    Wt[o * 128 + ki] = (bf16_t)s;
  }
  #pragma unroll
  for (int i = 0; i < 32; ++i) {            // A row b: [ x[b,n,0:64] | C1[n, b*64:b*64+64] ]
    int q  = tid + i * 256;
    int b  = q >> 7;
    int ki = q & 127;
    bf16_t v;
    if (ki < 64) v = (bf16_t)x[(size_t)b * NNODES * DIM + (size_t)n * DIM + ki];
    else         v = C1[(size_t)n * NNODES + b * 64 + (ki - 64)];
    At[b * 128 + ki] = v;
  }
  if (tid < 64) {
    float s = 0.f;
    #pragma unroll
    for (int d = 0; d < EMBED; ++d) s += e[d] * Bp[d * 64 + tid];
    biasv[tid] = s;
  }
  __syncthreads();

  const int lane = tid & 31, wave = tid >> 5;
  const int hs = (lane < 16) ? 0 : 1;
  const int l16 = lane & 15;
  const int a_k0 = hs * 16, b_k0 = hs * 16;

  #pragma unroll
  for (int t = 0; t < 2; ++t) {             // 16 output tiles / 8 waves
    int tile = wave * 2 + t;
    int mi = tile >> 2;                     // batch tile
    int ni = tile & 3;                      // output-channel tile
    f32x8 acc = {};
    #pragma unroll
    for (int kc = 0; kc < 4; ++kc) {
      int kk = kc * 32;
      bf16x16 a, w;
      {
        int row = mi * 16 + l16;
        const bf16_t* p = &At[row * 128 + kk + a_k0];
        bf16x8 lo = *(const bf16x8*)p;
        bf16x8 hi = *(const bf16x8*)(p + 8);
        #pragma unroll
        for (int u = 0; u < 8; ++u) { a[u] = lo[u]; a[u + 8] = hi[u]; }
      }
      {
        int col = ni * 16 + l16;
        w = *(const bf16x16*)(&Wt[col * 128 + kk + b_k0]);
      }
      acc = __builtin_amdgcn_wmma_f32_16x16x32_bf16(false, a, false, w,
                                                    (short)0, acc, false, false);
    }
    int o = ni * 16 + l16;
    float bv = biasv[o];
    #pragma unroll
    for (int r = 0; r < 8; ++r) {
      int b = mi * 16 + r + hs * 8;
      out[(size_t)b * NNODES * DIM + (size_t)n * DIM + o] = acc[r] + bv;
    }
  }
}

// ---------------------------------------------------------------------------
extern "C" void kernel_launch(void* const* d_in, const int* in_sizes, int n_in,
                              void* d_out, int out_size, void* d_ws, size_t ws_size,
                              hipStream_t stream) {
  const float* x  = (const float*)d_in[0];   // [64, 4096, 64]
  const float* E  = (const float*)d_in[1];   // [4096, 10]
  const float* Wp = (const float*)d_in[2];   // [10, 2, 64, 64]
  const float* Bp = (const float*)d_in[3];   // [10, 64]
  float* out = (float*)d_out;                // [64, 4096, 64]

  const size_t NN = (size_t)NNODES * NNODES;
  bf16_t* S  = (bf16_t*)d_ws;   // 32 MB: softmax(relu(E E^T)) in bf16
  bf16_t* XT = S + NN;          // 32 MB: x transposed to [b*64+c][m] bf16
  bf16_t* C1 = XT + NN;         // 32 MB: diffusion result S @ x, bf16

  k_build_xt<<<dim3(NBATCH, NNODES / 64), 256, 0, stream>>>(x, XT);
  k_softmax <<<dim3(NNODES),              256, 0, stream>>>(E, S);
  k_diffuse <<<dim3(NNODES / 128, NNODES / 128), 256, 0, stream>>>(S, XT, C1);
  k_final   <<<dim3(NNODES),              256, 0, stream>>>(x, E, Wp, Bp, C1, out);
}
